// NodeModel_67791763800206
// MI455X (gfx1250) — matrix operand order repaired and checked by reading, not compiled, not run on hardware
//
#include <hip/hip_runtime.h>
#include <hip/hip_bf16.h>

// ---------------------------------------------------------------------------
// Problem constants (from the reference)
// ---------------------------------------------------------------------------
#define NN 50000      // nodes
#define NE 150000     // edges
#define NG 64         // graphs
#define DN 9          // node feature dim
#define DE 512        // edge feature dim
#define DU 16         // graph feature dim
#define HD 512        // hidden
#define K1 544        // (9+512)=521 padded to 17*32
#define K2 544        // (9+512+16)=537 padded to 17*32

typedef __attribute__((ext_vector_type(16))) _Float16 v16h;
typedef __attribute__((ext_vector_type(8)))  _Float16 v8h;
typedef __attribute__((ext_vector_type(4)))  _Float16 v4h;
typedef __attribute__((ext_vector_type(8)))  float    v8f;
typedef __attribute__((ext_vector_type(4)))  float    v4f;

union V16 { v16h v; v8h h[2]; };

#define WMMA_F16(a, b, c) \
    __builtin_amdgcn_wmma_f32_16x16x32_f16(false, (a), false, (b), (short)0, (c), false, false)

// Workspace layout (bytes). sums and cnt are contiguous so one zero pass covers both.
static const size_t SUMS_OFF = 0;                                                   // float[NN*HD]
static const size_t CNT_OFF  = SUMS_OFF + (size_t)NN * HD * sizeof(float);          // float[NN]
static const size_t W1A_OFF  = CNT_OFF  + (size_t)NN * sizeof(float);               // f16[K1*HD]
static const size_t W1B_OFF  = W1A_OFF  + (size_t)K1 * HD * sizeof(_Float16);       // f16[HD*HD]
static const size_t W2A_OFF  = W1B_OFF  + (size_t)HD * HD * sizeof(_Float16);       // f16[K2*HD]

// ---------------------------------------------------------------------------
// Helpers
// ---------------------------------------------------------------------------
__global__ void zero_f32_kernel(float* __restrict__ p, size_t n) {
    size_t i = (size_t)blockIdx.x * blockDim.x + threadIdx.x;
    size_t stride = (size_t)gridDim.x * blockDim.x;
    for (; i < n; i += stride) p[i] = 0.0f;
}

// fp32 -> fp16 weight conversion with zero-padded extra K rows (same col count)
__global__ void cvt_pad_kernel(const float* __restrict__ src, _Float16* __restrict__ dst,
                               int rows_src, int rows_dst, int cols) {
    size_t total = (size_t)rows_dst * cols;
    size_t i = (size_t)blockIdx.x * blockDim.x + threadIdx.x;
    size_t stride = (size_t)gridDim.x * blockDim.x;
    for (; i < total; i += stride) {
        int r = (int)(i / cols);
        dst[i] = (r < rows_src) ? (_Float16)src[i] : (_Float16)0.0f;
    }
}

// per-destination-node edge counts (for the mean)
__global__ void count_kernel(const int* __restrict__ edge_row, float* __restrict__ cnt, int E) {
    int e = blockIdx.x * blockDim.x + threadIdx.x;
    if (e < E) atomicAdd(&cnt[edge_row[e]], 1.0f);
}

// ---------------------------------------------------------------------------
// Edge MLP: h = relu([x[col],ea] @ W1a + b1a) @ W1b + b1b ; scatter-add to sums
// One block = 32 edges x 512 cols. 8 waves; each wave owns a 32x64 slice as
// two 16-row halves sharing the same B fragments (32 FLOP / B-byte).
// ---------------------------------------------------------------------------
__global__ __launch_bounds__(256)
void edge_mlp_kernel(const float* __restrict__ x,
                     const int* __restrict__ edge_index,   // [2][NE]
                     const float* __restrict__ edge_attr,  // [NE][DE]
                     const _Float16* __restrict__ W1a,     // [K1][HD]
                     const float* __restrict__ b1a,
                     const _Float16* __restrict__ W1b,     // [HD][HD]
                     const float* __restrict__ b1b,
                     float* __restrict__ sums,             // [NN][HD]
                     int E)
{
    __shared__ _Float16 abuf[32 * 32];     // A staging tile (f16)
    __shared__ _Float16 hbuf[32 * 520];    // layer-1 activations (f16, padded stride)
    __shared__ int rowIdx[32];

    const int tid  = threadIdx.x;
    const int lane = tid & 31;
    const int wave = tid >> 5;
    const int e0   = blockIdx.x * 32;
    const int nbase = wave * 64;

    if (tid < 32) {
        int e = e0 + tid;
        rowIdx[tid] = (e < E) ? edge_index[e] : -1;   // destination node (row)
    }

    // ---------------- GEMM 1: K = 544 -------------------------------------
    v8f acc[2][4] = {{v8f{}, v8f{}, v8f{}, v8f{}}, {v8f{}, v8f{}, v8f{}, v8f{}}};
    for (int kt = 0; kt < K1 / 32; ++kt) {
        __syncthreads();
        // stage 32x32 A tile: each thread handles 4 contiguous K elements
        {
            int t  = tid * 4;
            int m  = t >> 5;
            int kk = t & 31;
            int k  = kt * 32 + kk;
            int e  = e0 + m;
            _Float16 h4[4];
            if (e < E && k >= DN && (k + 3) < DN + DE) {
                // fully inside edge_attr: one streamed (non-temporal) 16B load
                v4f f = __builtin_nontemporal_load(
                            (const v4f*)&edge_attr[(size_t)e * DE + (k - DN)]);
                h4[0] = (_Float16)f.x; h4[1] = (_Float16)f.y;
                h4[2] = (_Float16)f.z; h4[3] = (_Float16)f.w;
            } else {
                #pragma unroll
                for (int i = 0; i < 4; ++i) {
                    int kq = k + i;
                    float val = 0.0f;
                    if (e < E) {
                        if (kq < DN) {
                            int col = edge_index[E + e];   // source node
                            val = x[col * DN + kq];
                        } else if (kq < DN + DE) {
                            val = edge_attr[(size_t)e * DE + (kq - DN)];
                        }
                    }
                    h4[i] = (_Float16)val;
                }
            }
            *(v4h*)&abuf[m * 32 + kk] = *(v4h*)h4;
        }
        __syncthreads();

        // A fragments for both 16-row halves
        V16 a0, a1;
        {
            int ml = lane & 15;
            int base = (lane < 16) ? 0 : 8;
            a0.h[0] = *(const v8h*)&abuf[ml * 32 + base];
            a0.h[1] = *(const v8h*)&abuf[ml * 32 + 16 + base];
            a1.h[0] = *(const v8h*)&abuf[(ml + 16) * 32 + base];
            a1.h[1] = *(const v8h*)&abuf[(ml + 16) * 32 + 16 + base];
        }
        // B fragment: lane = N column (lane%16), elements = K in table order (8h+s, 16+8h+s-8)
        const _Float16* brow = &W1a[(size_t)(kt * 32 + ((lane < 16) ? 0 : 8)) * HD];
        #pragma unroll
        for (int t = 0; t < 4; ++t) {
            v16h b;
            #pragma unroll
            for (int s = 0; s < 16; ++s) b[s] = brow[(size_t)(s < 8 ? s : s + 8) * HD + nbase + t * 16 + (lane & 15)];
            acc[0][t] = WMMA_F16(a0.v, b, acc[0][t]);
            acc[1][t] = WMMA_F16(a1.v, b, acc[1][t]);
        }
    }

    // bias + relu + spill layer-1 activations to LDS (f16, row-major)
    {
        int nl   = lane & 15;
        int mofs = (lane < 16) ? 0 : 8;
        #pragma unroll
        for (int mg = 0; mg < 2; ++mg) {
            #pragma unroll
            for (int t = 0; t < 4; ++t) {
                int n = nbase + t * 16 + nl;
                float bias = b1a[n];
                #pragma unroll
                for (int vv = 0; vv < 8; ++vv) {
                    float f = acc[mg][t][vv] + bias;
                    f = f > 0.0f ? f : 0.0f;
                    hbuf[(mg * 16 + vv + mofs) * 520 + n] = (_Float16)f;
                }
            }
        }
    }
    __syncthreads();

    // ---------------- GEMM 2: K = 512, A from LDS -------------------------
    v8f acc2[2][4] = {{v8f{}, v8f{}, v8f{}, v8f{}}, {v8f{}, v8f{}, v8f{}, v8f{}}};
    for (int kt = 0; kt < HD / 32; ++kt) {
        V16 a0, a1;
        {
            int ml = lane & 15;
            int base = (lane < 16) ? 0 : 8;
            a0.h[0] = *(const v8h*)&hbuf[ml * 520 + kt * 32 + base];
            a0.h[1] = *(const v8h*)&hbuf[ml * 520 + kt * 32 + 16 + base];
            a1.h[0] = *(const v8h*)&hbuf[(ml + 16) * 520 + kt * 32 + base];
            a1.h[1] = *(const v8h*)&hbuf[(ml + 16) * 520 + kt * 32 + 16 + base];
        }
        const _Float16* brow = &W1b[(size_t)(kt * 32 + ((lane < 16) ? 0 : 8)) * HD];
        #pragma unroll
        for (int t = 0; t < 4; ++t) {
            v16h b;
            #pragma unroll
            for (int s = 0; s < 16; ++s) b[s] = brow[(size_t)(s < 8 ? s : s + 8) * HD + nbase + t * 16 + (lane & 15)];
            acc2[0][t] = WMMA_F16(a0.v, b, acc2[0][t]);
            acc2[1][t] = WMMA_F16(a1.v, b, acc2[1][t]);
        }
    }

    // bias + scatter-add to per-node sums
    {
        int nl   = lane & 15;
        int mofs = (lane < 16) ? 0 : 8;
        #pragma unroll
        for (int mg = 0; mg < 2; ++mg) {
            #pragma unroll
            for (int t = 0; t < 4; ++t) {
                int n = nbase + t * 16 + nl;
                float bias = b1b[n];
                #pragma unroll
                for (int vv = 0; vv < 8; ++vv) {
                    int r = rowIdx[mg * 16 + vv + mofs];
                    if (r >= 0) {
                        atomicAdd(&sums[(size_t)r * HD + n], acc2[mg][t][vv] + bias);
                    }
                }
            }
        }
    }
}

// ---------------------------------------------------------------------------
// Node MLP: out = relu([x, sums/cnt, u[batch]] @ W2a + b2a) @ W2b + b2b
// One block = 32 nodes. Layer 2 (512 -> 1) done with LDS f32 atomics.
// ---------------------------------------------------------------------------
__global__ __launch_bounds__(256)
void node_mlp_kernel(const float* __restrict__ x,
                     const float* __restrict__ u,          // [NG][DU]
                     const int* __restrict__ batch,        // [NN]
                     const float* __restrict__ sums,       // [NN][HD]
                     const float* __restrict__ cnt,        // [NN]
                     const _Float16* __restrict__ W2a,     // [K2][HD]
                     const float* __restrict__ b2a,
                     const float* __restrict__ W2b,        // [HD]
                     const float* __restrict__ b2b,        // [1]
                     float* __restrict__ out, int N)
{
    __shared__ _Float16 abuf[32 * 32];
    __shared__ float invc[32];
    __shared__ float oacc[32];

    const int tid  = threadIdx.x;
    const int lane = tid & 31;
    const int wave = tid >> 5;
    const int n0   = blockIdx.x * 32;
    const int nbase = wave * 64;

    if (tid < 32) {
        int nd = n0 + tid;
        float c = (nd < N) ? cnt[nd] : 1.0f;
        invc[tid] = 1.0f / fmaxf(c, 1.0f);
        oacc[tid] = 0.0f;
    }

    v8f acc[2][4] = {{v8f{}, v8f{}, v8f{}, v8f{}}, {v8f{}, v8f{}, v8f{}, v8f{}}};
    for (int kt = 0; kt < K2 / 32; ++kt) {
        __syncthreads();
        // stage 32x32 tile of z = [x, agg, u[batch]] (4 elements per thread)
        {
            int t  = tid * 4;
            int m  = t >> 5;
            int kk = t & 31;
            int nd = n0 + m;
            _Float16 h4[4];
            #pragma unroll
            for (int i = 0; i < 4; ++i) {
                int k = kt * 32 + kk + i;
                float val = 0.0f;
                if (nd < N) {
                    if (k < DN) {
                        val = x[nd * DN + k];
                    } else if (k < DN + HD) {
                        val = sums[(size_t)nd * HD + (k - DN)] * invc[m];
                    } else if (k < DN + HD + DU) {
                        int g = batch[nd];
                        val = u[g * DU + (k - DN - HD)];
                    }
                }
                h4[i] = (_Float16)val;
            }
            *(v4h*)&abuf[m * 32 + kk] = *(v4h*)h4;
        }
        __syncthreads();

        V16 a0, a1;
        {
            int ml = lane & 15;
            int base = (lane < 16) ? 0 : 8;
            a0.h[0] = *(const v8h*)&abuf[ml * 32 + base];
            a0.h[1] = *(const v8h*)&abuf[ml * 32 + 16 + base];
            a1.h[0] = *(const v8h*)&abuf[(ml + 16) * 32 + base];
            a1.h[1] = *(const v8h*)&abuf[(ml + 16) * 32 + 16 + base];
        }
        const _Float16* brow = &W2a[(size_t)(kt * 32 + ((lane < 16) ? 0 : 8)) * HD];
        #pragma unroll
        for (int t = 0; t < 4; ++t) {
            v16h b;
            #pragma unroll
            for (int s = 0; s < 16; ++s) b[s] = brow[(size_t)(s < 8 ? s : s + 8) * HD + nbase + t * 16 + (lane & 15)];
            acc[0][t] = WMMA_F16(a0.v, b, acc[0][t]);
            acc[1][t] = WMMA_F16(a1.v, b, acc[1][t]);
        }
    }

    // layer 2: relu(.) dot W2b, reduce over n via LDS atomics
    {
        int nl   = lane & 15;
        int mofs = (lane < 16) ? 0 : 8;
        #pragma unroll
        for (int mg = 0; mg < 2; ++mg) {
            #pragma unroll
            for (int t = 0; t < 4; ++t) {
                int n = nbase + t * 16 + nl;
                float bias = b2a[n];
                float w = W2b[n];
                #pragma unroll
                for (int vv = 0; vv < 8; ++vv) {
                    float f = acc[mg][t][vv] + bias;
                    f = f > 0.0f ? f : 0.0f;
                    atomicAdd(&oacc[mg * 16 + vv + mofs], f * w);
                }
            }
        }
    }
    __syncthreads();
    if (tid < 32) {
        int nd = n0 + tid;
        if (nd < N) out[nd] = oacc[tid] + b2b[0];
    }
}

// ---------------------------------------------------------------------------
// Launch
// ---------------------------------------------------------------------------
extern "C" void kernel_launch(void* const* d_in, const int* in_sizes, int n_in,
                              void* d_out, int out_size, void* d_ws, size_t ws_size,
                              hipStream_t stream) {
    (void)in_sizes; (void)n_in; (void)out_size; (void)ws_size;

    const float* x          = (const float*)d_in[0];
    const int*   edge_index = (const int*)d_in[1];     // [2][NE]
    const float* edge_attr  = (const float*)d_in[2];
    const float* u          = (const float*)d_in[3];
    const int*   batch      = (const int*)d_in[4];
    const float* W1a        = (const float*)d_in[5];
    const float* b1a        = (const float*)d_in[6];
    const float* W1b        = (const float*)d_in[7];
    const float* b1b        = (const float*)d_in[8];
    const float* W2a        = (const float*)d_in[9];
    const float* b2a        = (const float*)d_in[10];
    const float* W2b        = (const float*)d_in[11];
    const float* b2b        = (const float*)d_in[12];
    float* out = (float*)d_out;

    char* ws = (char*)d_ws;
    float*    sums  = (float*)(ws + SUMS_OFF);
    float*    cnt   = (float*)(ws + CNT_OFF);
    _Float16* W1a_h = (_Float16*)(ws + W1A_OFF);
    _Float16* W1b_h = (_Float16*)(ws + W1B_OFF);
    _Float16* W2a_h = (_Float16*)(ws + W2A_OFF);

    // 1) zero sums + cnt (contiguous)
    zero_f32_kernel<<<2048, 256, 0, stream>>>(sums, (size_t)NN * HD + NN);

    // 2) convert + pad weights to f16
    cvt_pad_kernel<<<512, 256, 0, stream>>>(W1a, W1a_h, DN + DE, K1, HD);
    cvt_pad_kernel<<<512, 256, 0, stream>>>(W1b, W1b_h, HD, HD, HD);
    cvt_pad_kernel<<<512, 256, 0, stream>>>(W2a, W2a_h, DN + HD + DU, K2, HD);

    // 3) destination counts
    count_kernel<<<(NE + 255) / 256, 256, 0, stream>>>(edge_index, cnt, NE);

    // 4) edge MLP + scatter-add
    edge_mlp_kernel<<<(NE + 31) / 32, 256, 0, stream>>>(
        x, edge_index, edge_attr, W1a_h, b1a, W1b_h, b1b, sums, NE);

    // 5) node MLP
    node_mlp_kernel<<<(NN + 31) / 32, 256, 0, stream>>>(
        x, u, batch, sums, cnt, W2a_h, b2a, W2b, b2b, out, NN);
}
